// GeometricPositionUpdate_10393820857006
// MI455X (gfx1250) — matrix operand order repaired and checked by reading, not compiled, not run on hardware
//
#include <hip/hip_runtime.h>

// ---------------------------------------------------------------------------
// Problem constants (fixed by setup_inputs: B=4, N=2048, D=1024, K=8)
// ---------------------------------------------------------------------------
#define BSZ   4
#define NPT   2048
#define DIM   1024
#define KNN   8
#define MROWS (BSZ * NPT)          // 8192  (b,i) rows
#define HROWS (MROWS * KNN)        // 65536 (b,i,j) rows

typedef __attribute__((ext_vector_type(16))) __bf16 v16bf;
typedef __attribute__((ext_vector_type(8)))  float  v8f;
typedef __attribute__((ext_vector_type(4)))  int    v4i;

union FragBF { v16bf v; uint4 u[2]; };
union HPack  { __bf16 h[16]; uint4 u[2]; };

// Tiling: block = 128x128 output, K-chunk 32, 8 waves each doing 64x32.
#define TM 128
#define TN 128
#define TK 32
#define NS (DIM / TK)   // 32 K-stages
#define LROW 40         // 32 + 8 bf16 pad -> 80B row stride (16B aligned, bank-spread)

// ---------------------------------------------------------------------------
// Async global->LDS path (CDNA5 GLOBAL_LOAD_ASYNC_TO_LDS_B128, ASYNCcnt).
// Signature (from hipcc diagnostics): arg0 = v4i in AS1 (global), arg1 = v4i
// in AS3 (LDS), then imm offset + imm cpol.
// Guarded: falls back to global->VGPR->ds_store if the builtins are absent.
// ---------------------------------------------------------------------------
#if __has_builtin(__builtin_amdgcn_global_load_async_to_lds_b128) && \
    __has_builtin(__builtin_amdgcn_s_wait_asynccnt)
#define USE_ASYNC_LDS 1
typedef __attribute__((address_space(1))) v4i gv4i;
typedef __attribute__((address_space(3))) v4i lv4i;
#else
#define USE_ASYNC_LDS 0
#endif

// Copy 16 bf16 (32B) from global to LDS.
__device__ __forceinline__ void copy16(__bf16* dst, const __bf16* src) {
#if USE_ASYNC_LDS
  __builtin_amdgcn_global_load_async_to_lds_b128((gv4i*)src, (lv4i*)dst, 0, 0);
  __builtin_amdgcn_global_load_async_to_lds_b128((gv4i*)(src + 8), (lv4i*)(dst + 8), 0, 0);
#else
  const uint4 a = *(const uint4*)src;
  const uint4 b = *(const uint4*)(src + 8);
  *(uint4*)dst = a;
  *(uint4*)(dst + 8) = b;
#endif
}

// ---------------------------------------------------------------------------
// KNN: one block = 256 queries of one batch; whole batch's pos staged in LDS.
// ---------------------------------------------------------------------------
__global__ __launch_bounds__(256) void knn_kernel(const float* __restrict__ pos,
                                                  int* __restrict__ idx) {
  __shared__ float ps[NPT * 3];
  const int b = blockIdx.y;
  const int tid = threadIdx.x;
  const float* pb = pos + (size_t)b * NPT * 3;
  for (int i = tid; i < NPT * 3; i += 256) ps[i] = pb[i];
  __syncthreads();

  const int q = blockIdx.x * 256 + tid;
  const float qx = ps[q * 3 + 0], qy = ps[q * 3 + 1], qz = ps[q * 3 + 2];

  float bd[KNN];
  int   bi[KNN];
#pragma unroll
  for (int j = 0; j < KNN; ++j) { bd[j] = 3.4e38f; bi[j] = 0; }

  for (int p = 0; p < NPT; ++p) {
    const float dx = qx - ps[p * 3 + 0];
    const float dy = qy - ps[p * 3 + 1];
    const float dz = qz - ps[p * 3 + 2];
    const float d2 = dx * dx + dy * dy + dz * dz;
    if (d2 < bd[KNN - 1]) {
      bd[KNN - 1] = d2; bi[KNN - 1] = p;
#pragma unroll
      for (int j = KNN - 1; j >= 1; --j) {
        if (bd[j] < bd[j - 1]) {
          float td = bd[j]; bd[j] = bd[j - 1]; bd[j - 1] = td;
          int   ti = bi[j]; bi[j] = bi[j - 1]; bi[j - 1] = ti;
        }
      }
    }
  }
  int* o = idx + ((size_t)b * NPT + q) * KNN;
#pragma unroll
  for (int j = 0; j < KNN; ++j) o[j] = bi[j];
}

// ---------------------------------------------------------------------------
// fp32 -> bf16 elementwise
// ---------------------------------------------------------------------------
__global__ __launch_bounds__(256) void cvt_bf16_kernel(const float* __restrict__ src,
                                                       __bf16* __restrict__ dst, int n) {
  int i = blockIdx.x * 256 + threadIdx.x;
  if (i < n) dst[i] = (__bf16)src[i];
}

// src [R][C] f32 -> dst [C][R] bf16  (weights in N-major for contiguous B-frags)
__global__ __launch_bounds__(256) void transpose_cvt_kernel(const float* __restrict__ src,
                                                            __bf16* __restrict__ dst,
                                                            int R, int C) {
  int o = blockIdx.x * 256 + threadIdx.x;
  if (o < R * C) {
    int r = o % R;     // src row  (contraction dim)
    int c = o / R;     // src col  (output dim) = dst row
    dst[o] = (__bf16)src[(size_t)r * C + c];
  }
}

// ---------------------------------------------------------------------------
// GEMM1: Y{1,2}[8192][1024] = Xb[8192][1024] @ W1t[1024][2048] slice
//   blockIdx.z = 0 -> W1 top half (koff 0)    -> Y1
//   blockIdx.z = 1 -> W1 bottom half (koff D) -> Y2
// Double-buffered LDS; tiles streamed with async global->LDS (ASYNCcnt).
// ---------------------------------------------------------------------------
__global__ __launch_bounds__(256) void gemm1_kernel(const __bf16* __restrict__ A,
                                                    const __bf16* __restrict__ Bt,
                                                    float* __restrict__ Y1,
                                                    float* __restrict__ Y2) {
  __shared__ __bf16 As[2][TM][LROW];
  __shared__ __bf16 Bs[2][TN][LROW];

  const int tid   = threadIdx.x;
  const int Mbase = blockIdx.y * TM;
  const int Nbase = blockIdx.x * TN;
  const int koff  = blockIdx.z ? DIM : 0;
  float* Y = blockIdx.z ? Y2 : Y1;

  const int wave = tid >> 5, lane = tid & 31;
  const int wm = wave >> 2, wn = wave & 3;     // 2x4 wave grid
  const int hh = lane >> 4, ln = lane & 15;

  v8f acc[4][2];
  const v8f z8 = {0.f, 0.f, 0.f, 0.f, 0.f, 0.f, 0.f, 0.f};
#pragma unroll
  for (int mt = 0; mt < 4; ++mt)
#pragma unroll
    for (int nt = 0; nt < 2; ++nt) acc[mt][nt] = z8;

  const int rowX = tid >> 1;          // 0..127 staged row
  const int kh   = (tid & 1) * 16;    // 0 / 16
  const __bf16* gA = A  + (size_t)(Mbase + rowX) * DIM + kh;
  const __bf16* gB = Bt + (size_t)(Nbase + rowX) * (2 * DIM) + koff + kh;

  // Issue stage s into buffer s&1 (4 async b128 per thread).
  auto load_stage = [&](int s) {
    const int p = s & 1, kc = s * TK;
    copy16(&As[p][rowX][kh], gA + kc);
    copy16(&Bs[p][rowX][kh], gB + kc);
  };

  load_stage(0);
  for (int s = 0; s < NS; ++s) {
    if (s + 1 < NS) load_stage(s + 1);
#if USE_ASYNC_LDS
    if (s + 1 < NS) __builtin_amdgcn_s_wait_asynccnt(4);  // stage s retired
    else            __builtin_amdgcn_s_wait_asynccnt(0);
#endif
    __syncthreads();

    const __bf16 (*Ab)[LROW] = As[s & 1];
    const __bf16 (*Bb)[LROW] = Bs[s & 1];
    FragBF fa[4], fb[2];
#pragma unroll
    for (int mt = 0; mt < 4; ++mt) {
      const int m = 64 * wm + 16 * mt + ln;
      fa[mt].u[0] = *(const uint4*)&Ab[m][8 * hh];
      fa[mt].u[1] = *(const uint4*)&Ab[m][16 + 8 * hh];
    }
#pragma unroll
    for (int nt = 0; nt < 2; ++nt) {
      const int n = 32 * wn + 16 * nt + ln;
      fb[nt].u[0] = *(const uint4*)&Bb[n][8 * hh];
      fb[nt].u[1] = *(const uint4*)&Bb[n][16 + 8 * hh];
    }
#pragma unroll
    for (int mt = 0; mt < 4; ++mt)
#pragma unroll
      for (int nt = 0; nt < 2; ++nt)
        acc[mt][nt] = __builtin_amdgcn_wmma_f32_16x16x32_bf16(
            false, fa[mt].v, false, fb[nt].v, (short)0, acc[mt][nt], false, false);
    __syncthreads();   // all waves done reading buf before it is re-filled
  }

#pragma unroll
  for (int mt = 0; mt < 4; ++mt)
#pragma unroll
    for (int nt = 0; nt < 2; ++nt) {
      const int col  = Nbase + 32 * wn + 16 * nt + ln;
      const int row0 = Mbase + 64 * wm + 16 * mt + 8 * hh;
#pragma unroll
      for (int r = 0; r < 8; ++r)
        Y[(size_t)(row0 + r) * DIM + col] = acc[mt][nt][r];
    }
}

// ---------------------------------------------------------------------------
// GEMM2 fused: H[65536][1024] built on the fly (gather + relu + bf16 cvt),
// Z = H @ W2t^T, epilogue max over each 8-row (b,i) group + b2 -> out.
// W2t tile streamed async->LDS; H tile computed in VGPRs -> ds_store.
// ---------------------------------------------------------------------------
__global__ __launch_bounds__(256) void gemm2_kernel(const float* __restrict__ Y1,
                                                    const float* __restrict__ Y2,
                                                    const int* __restrict__ idx,
                                                    const __bf16* __restrict__ W2t,
                                                    const float* __restrict__ b1,
                                                    const float* __restrict__ b2,
                                                    float* __restrict__ out) {
  __shared__ __bf16 Hs[2][TM][LROW];
  __shared__ __bf16 Bs[2][TN][LROW];

  const int tid   = threadIdx.x;
  const int Mbase = blockIdx.y * TM;     // H-row space (65536)
  const int Nbase = blockIdx.x * TN;

  const int wave = tid >> 5, lane = tid & 31;
  const int wm = wave >> 2, wn = wave & 3;
  const int hh = lane >> 4, ln = lane & 15;

  v8f acc[4][2];
  const v8f z8 = {0.f, 0.f, 0.f, 0.f, 0.f, 0.f, 0.f, 0.f};
#pragma unroll
  for (int mt = 0; mt < 4; ++mt)
#pragma unroll
    for (int nt = 0; nt < 2; ++nt) acc[mt][nt] = z8;

  // Per-thread staged H row: resolve gather once.
  const int rowX = tid >> 1;
  const int kh   = (tid & 1) * 16;
  const int gm   = Mbase + rowX;      // global H row
  const int biq  = gm >> 3;           // (b*N + i)
  const int jj   = gm & 7;
  const int bb   = biq >> 11;         // / N
  const int nbr  = idx[biq * KNN + jj];
  const float*  y1row = Y1 + (size_t)biq * DIM;
  const float*  y2row = Y2 + (size_t)(bb * NPT + nbr) * DIM;
  const __bf16* gB    = W2t + (size_t)(Nbase + rowX) * DIM + kh;

  // Stage s: async-stream W2t tile; build H tile (gather+bias+relu+bf16).
  auto build_stage = [&](int s) {
    const int p = s & 1, kc = s * TK, k0 = kc + kh;
    copy16(&Bs[p][rowX][kh], gB + kc);   // 2 async b128
    float4 P[4], Q[4], Cc[4];
#pragma unroll
    for (int t = 0; t < 4; ++t) {
      P[t]  = *(const float4*)(y1row + k0 + 4 * t);
      Q[t]  = *(const float4*)(y2row + k0 + 4 * t);
      Cc[t] = *(const float4*)(b1 + k0 + 4 * t);
    }
    HPack hv;
#pragma unroll
    for (int t = 0; t < 4; ++t) {
      hv.h[4 * t + 0] = (__bf16)fmaxf(P[t].x + Q[t].x + Cc[t].x, 0.0f);
      hv.h[4 * t + 1] = (__bf16)fmaxf(P[t].y + Q[t].y + Cc[t].y, 0.0f);
      hv.h[4 * t + 2] = (__bf16)fmaxf(P[t].z + Q[t].z + Cc[t].z, 0.0f);
      hv.h[4 * t + 3] = (__bf16)fmaxf(P[t].w + Q[t].w + Cc[t].w, 0.0f);
    }
    *(uint4*)&Hs[p][rowX][kh]     = hv.u[0];
    *(uint4*)&Hs[p][rowX][kh + 8] = hv.u[1];
  };

  build_stage(0);
  for (int s = 0; s < NS; ++s) {
    if (s + 1 < NS) build_stage(s + 1);
#if USE_ASYNC_LDS
    if (s + 1 < NS) __builtin_amdgcn_s_wait_asynccnt(2);  // stage s B-tile retired
    else            __builtin_amdgcn_s_wait_asynccnt(0);
#endif
    __syncthreads();

    const __bf16 (*Ab)[LROW] = Hs[s & 1];
    const __bf16 (*Bb)[LROW] = Bs[s & 1];
    FragBF fa[4], fb[2];
#pragma unroll
    for (int mt = 0; mt < 4; ++mt) {
      const int m = 64 * wm + 16 * mt + ln;
      fa[mt].u[0] = *(const uint4*)&Ab[m][8 * hh];
      fa[mt].u[1] = *(const uint4*)&Ab[m][16 + 8 * hh];
    }
#pragma unroll
    for (int nt = 0; nt < 2; ++nt) {
      const int n = 32 * wn + 16 * nt + ln;
      fb[nt].u[0] = *(const uint4*)&Bb[n][8 * hh];
      fb[nt].u[1] = *(const uint4*)&Bb[n][16 + 8 * hh];
    }
#pragma unroll
    for (int mt = 0; mt < 4; ++mt)
#pragma unroll
      for (int nt = 0; nt < 2; ++nt)
        acc[mt][nt] = __builtin_amdgcn_wmma_f32_16x16x32_bf16(
            false, fa[mt].v, false, fb[nt].v, (short)0, acc[mt][nt], false, false);
    __syncthreads();
  }

  // Epilogue: each lane's 8 accum VGPRs == one full j-group (rows 8h..8h+7)
  // of one column -> per-lane max, + b2, one store per 16x16 tile.
#pragma unroll
  for (int mt = 0; mt < 4; ++mt)
#pragma unroll
    for (int nt = 0; nt < 2; ++nt) {
      const v8f a = acc[mt][nt];
      float mx = fmaxf(fmaxf(fmaxf(a[0], a[1]), fmaxf(a[2], a[3])),
                       fmaxf(fmaxf(a[4], a[5]), fmaxf(a[6], a[7])));
      const int col  = Nbase + 32 * wn + 16 * nt + ln;
      const int row0 = Mbase + 64 * wm + 16 * mt + 8 * hh;   // multiple of 8
      out[(size_t)(row0 >> 3) * DIM + col] = mx + b2[col];
    }
}

// ---------------------------------------------------------------------------
// Launch
// ---------------------------------------------------------------------------
extern "C" void kernel_launch(void* const* d_in, const int* in_sizes, int n_in,
                              void* d_out, int out_size, void* d_ws, size_t ws_size,
                              hipStream_t stream) {
  const float* x   = (const float*)d_in[0];
  const float* pos = (const float*)d_in[1];
  const float* W1  = (const float*)d_in[2];
  const float* b1  = (const float*)d_in[3];
  const float* W2  = (const float*)d_in[4];
  const float* b2  = (const float*)d_in[5];
  float* out = (float*)d_out;

  char* ws = (char*)d_ws;
  size_t off = 0;
  int*    idxb = (int*)(ws + off);    off += (size_t)HROWS * sizeof(int);          // 256 KB
  __bf16* Xb   = (__bf16*)(ws + off); off += (size_t)MROWS * DIM * 2;              // 16 MB
  __bf16* W1t  = (__bf16*)(ws + off); off += (size_t)DIM * 2 * DIM * 2;            // 4 MB
  __bf16* W2t  = (__bf16*)(ws + off); off += (size_t)DIM * DIM * 2;                // 2 MB
  float*  Y1   = (float*)(ws + off);  off += (size_t)MROWS * DIM * sizeof(float);  // 32 MB
  float*  Y2   = (float*)(ws + off);  off += (size_t)MROWS * DIM * sizeof(float);  // 32 MB

  // 1) KNN indices
  knn_kernel<<<dim3(NPT / 256, BSZ), 256, 0, stream>>>(pos, idxb);

  // 2) Precision staging
  const int nx = MROWS * DIM;
  cvt_bf16_kernel<<<(nx + 255) / 256, 256, 0, stream>>>(x, Xb, nx);
  transpose_cvt_kernel<<<(2 * DIM * DIM + 255) / 256, 256, 0, stream>>>(W1, W1t, 2 * DIM, DIM);
  transpose_cvt_kernel<<<(DIM * DIM + 255) / 256, 256, 0, stream>>>(W2, W2t, DIM, DIM);

  // 3) Y1 = Xb @ W1_top, Y2 = Xb @ W1_bot   (z selects half)
  gemm1_kernel<<<dim3(DIM / TN, MROWS / TM, 2), 256, 0, stream>>>(Xb, W1t, Y1, Y2);

  // 4) Fused gather+relu GEMM2 with max-over-k epilogue
  gemm2_kernel<<<dim3(DIM / TN, HROWS / TM), 256, 0, stream>>>(Y1, Y2, idxb, W2t, b1, b2, out);
}
